// KMIPAttention_7851200217704
// MI455X (gfx1250) — compile-verified, matrix operand order checked by reading, not run on hardware
//
#include <hip/hip_runtime.h>
#include <hip/hip_bf16.h>

// ---------------------------------------------------------------------------
// KMIP (top-k) attention for MI455X / gfx1250, wave32, WMMA bf16 path.
//   B=4, S=4096, D=256, TOPK=32
// Kernel 0: pre-pack Wq/Wk/Wv (f32) into per-lane WMMA B-fragment bf16 layout.
// Kernel 1: QKV projection (bf16 WMMA, f32 accum + bias). Q,K -> bf16, V -> f32.
// Kernel 2: fused QK^T (bf16 WMMA) -> LDS score panel -> per-thread top-32 ->
//           merge -> softmax -> gather V -> out. Never materializes [B,S,S].
// ---------------------------------------------------------------------------

typedef __attribute__((ext_vector_type(16))) __bf16 v16bf;
typedef __attribute__((ext_vector_type(8)))  float  v8f;

union Frag16 { uint4 u[2]; v16bf v; };   // 32 bytes = 16 bf16

#define BATCH  4
#define SEQ    4096
#define DIM    256
#define TOPK   32
#define CHUNK  256          // keys per LDS score panel
#define NCHUNK (SEQ / CHUNK)

// Packed weight layout: fragment for (ct, kt, lane) at
//   Wp[((ct*8 + kt)*32 + lane) * 16 .. +15]
// holding, per the 16-bit B layout, K runs {kbase+8*kh+0..7} and
// {kbase+16+8*kh+0..7} of column (ct*16 + lane%16).
#define WPACK_FRAGS (16 * 8 * 32)        // 4096 fragments of 16 bf16

// ---------------------------------------------------------------------------
// Kernel 0: pack one f32 weight matrix into B-fragment-ordered bf16.
// One thread per fragment (4096 threads): 16 strided loads + converts, once.
// ---------------------------------------------------------------------------
__global__ __launch_bounds__(256)
void kmip_pack_w_kernel(const float* __restrict__ W, __bf16* __restrict__ Wp)
{
    const int id   = blockIdx.x * blockDim.x + threadIdx.x;  // 0..4095
    const int lane = id & 31;
    const int kt   = (id >> 5) & 7;
    const int ct   = id >> 8;
    const int col  = ct * 16 + (lane & 15);
    const int kh   = lane >> 4;
    const int kbase = kt * 32;

    __bf16* dst = Wp + (size_t)id * 16;
    #pragma unroll
    for (int j = 0; j < 8; ++j) {
        dst[j]     = (__bf16)W[(size_t)(kbase + 8 * kh + j) * DIM + col];
        dst[8 + j] = (__bf16)W[(size_t)(kbase + 16 + 8 * kh + j) * DIM + col];
    }
}

// ---------------------------------------------------------------------------
// Kernel 1: y = x @ W + b for W in {Wq,Wk,Wv}. One wave per 16x16 tile.
// A-fragment (16-bit A, 16x32): lanes 0-15 -> M=0..15, K in {0..7,16..23};
// lanes 16-31 same M, K in {8..15,24..31}. Contiguous 8-element K runs.
// B fragments come pre-packed: two b128 loads each, no conversion.
// ---------------------------------------------------------------------------
__global__ __launch_bounds__(128)
void kmip_qkv_kernel(const float* __restrict__ x,
                     const __bf16* __restrict__ Wqp, const float* __restrict__ bq,
                     const __bf16* __restrict__ Wkp, const float* __restrict__ bk,
                     const __bf16* __restrict__ Wvp, const float* __restrict__ bv,
                     __bf16* __restrict__ qb, __bf16* __restrict__ kb,
                     float* __restrict__ vf)
{
    const int lane = threadIdx.x & 31;
    const int wave = threadIdx.x >> 5;
    const int tile = blockIdx.x * 4 + wave;      // 16384 tiles total
    const int rt   = tile >> 4;                  // row tile (0..1023)
    const int ct   = tile & 15;                  // col tile (0..15)
    const int m    = lane & 15;
    const int kh   = lane >> 4;                  // K-half select
    const int arow = rt * 16 + m;
    const int col  = ct * 16 + (lane & 15);

    v8f accq = {}, acck = {}, accv = {};

    for (int kt = 0; kt < 8; ++kt) {
        const int kbase = kt * 32;

        // ---- A fragment from x (f32 -> bf16), two 8-float runs ----
        v16bf a;
        {
            const float* xp = x + (size_t)arow * DIM + kbase + 8 * kh;
            #pragma unroll
            for (int j = 0; j < 8; ++j) a[j]     = (__bf16)xp[j];
            #pragma unroll
            for (int j = 0; j < 8; ++j) a[8 + j] = (__bf16)xp[16 + j];
        }

        // ---- B fragments: pre-packed, 32B each ----
        const size_t fo = ((size_t)(ct * 8 + kt) * 32 + lane) * 16;
        Frag16 uq, uk, uv;
        uq.u[0] = *(const uint4*)(Wqp + fo);
        uq.u[1] = *(const uint4*)(Wqp + fo + 8);
        uk.u[0] = *(const uint4*)(Wkp + fo);
        uk.u[1] = *(const uint4*)(Wkp + fo + 8);
        uv.u[0] = *(const uint4*)(Wvp + fo);
        uv.u[1] = *(const uint4*)(Wvp + fo + 8);

        accq = __builtin_amdgcn_wmma_f32_16x16x32_bf16(false, a, false, uq.v,
                                                       (short)0, accq, false, false);
        acck = __builtin_amdgcn_wmma_f32_16x16x32_bf16(false, a, false, uk.v,
                                                       (short)0, acck, false, false);
        accv = __builtin_amdgcn_wmma_f32_16x16x32_bf16(false, a, false, uv.v,
                                                       (short)0, accv, false, false);
    }

    // C/D layout: VGPR r, lanes 0-15 -> M=r, lanes 16-31 -> M=r+8, N=lane%16.
    const float biasq = bq[col], biask = bk[col], biasv = bv[col];
    #pragma unroll
    for (int r = 0; r < 8; ++r) {
        const size_t orow = (size_t)(rt * 16 + r + 8 * kh) * DIM + col;
        qb[orow] = (__bf16)(accq[r] + biasq);
        kb[orow] = (__bf16)(acck[r] + biask);
        vf[orow] = accv[r] + biasv;
    }
}

// ---------------------------------------------------------------------------
// Kernel 2: fused scores + top-32 + softmax + V gather.
// One 128-thread block (4 waves) per (batch, 16-query tile).
// ---------------------------------------------------------------------------
__global__ __launch_bounds__(128)
void kmip_attn_kernel(const __bf16* __restrict__ qb,
                      const __bf16* __restrict__ kb,
                      const float*  __restrict__ vf,
                      float*        __restrict__ out)
{
    __shared__ float s_sc[16][CHUNK];       // 16 KB score panel
    __shared__ float s_mv[16][8][TOPK];     // 16 KB candidate values
    __shared__ int   s_mi[16][8][TOPK];     // 16 KB candidate indices
    __shared__ float s_p [16][TOPK];        //  2 KB softmax probs
    __shared__ int   s_pi[16][TOPK];        //  2 KB selected rows

    const int t    = threadIdx.x;
    const int lane = t & 31;
    const int wave = t >> 5;
    const int bat  = blockIdx.x >> 8;        // batch 0..3
    const int qt   = blockIdx.x & 255;       // query tile in batch
    const int q0   = bat * SEQ + qt * 16;    // first global query row

    const int mn = lane & 15;
    const int kh = lane >> 4;

    // Preload the 8 A-fragments (16 query rows x full D=256, bf16).
    v16bf afr[8];
    {
        const __bf16* qp = qb + (size_t)(q0 + mn) * DIM;
        #pragma unroll
        for (int kt = 0; kt < 8; ++kt) {
            const __bf16* p0 = qp + kt * 32 + 8 * kh;
            Frag16 u;
            u.u[0] = *(const uint4*)(p0);
            u.u[1] = *(const uint4*)(p0 + 16);
            afr[kt] = u.v;
        }
    }

    // Per-thread running top-32 (insertion-sorted descending, static indices
    // only -> stays in VGPRs). Thread t owns query (t>>3), key segment (t&7).
    float tv[TOPK]; int ti[TOPK];
    #pragma unroll
    for (int j = 0; j < TOPK; ++j) { tv[j] = -3.0e38f; ti[j] = 0; }
    const int sq = t >> 3;
    const int sg = t & 7;

    for (int ch = 0; ch < NCHUNK; ++ch) {
        const int keybase = ch * CHUNK;

        // ---- phase 1: 16x256 score panel via WMMA, panel -> LDS ----
        for (int tt = wave; tt < CHUNK / 16; tt += 4) {
            const int keyrow = bat * SEQ + keybase + tt * 16 + (lane & 15);
            const __bf16* kp = kb + (size_t)keyrow * DIM;
            v8f acc = {};
            #pragma unroll
            for (int kt = 0; kt < 8; ++kt) {
                const __bf16* p0 = kp + kt * 32 + 8 * kh;
                Frag16 u;
                u.u[0] = *(const uint4*)(p0);
                u.u[1] = *(const uint4*)(p0 + 16);
                acc = __builtin_amdgcn_wmma_f32_16x16x32_bf16(false, afr[kt], false, u.v,
                                                              (short)0, acc, false, false);
            }
            #pragma unroll
            for (int r = 0; r < 8; ++r)
                s_sc[r + 8 * kh][tt * 16 + (lane & 15)] = acc[r];
        }
        __syncthreads();

        // Pull the next chunk's K rows toward the caches while the VALU-heavy
        // selection pass runs (gfx1250 global_prefetch_b8 path).
        if (ch + 1 < NCHUNK) {
            const int nrow0 = bat * SEQ + (ch + 1) * CHUNK;
            __builtin_prefetch(kb + (size_t)(nrow0 + 2 * t)     * DIM, 0, 0);
            __builtin_prefetch(kb + (size_t)(nrow0 + 2 * t + 1) * DIM, 0, 0);
        }

        // ---- phase 2: scan 32 keys of my segment, keep running top-32 ----
        {
            const int kb0 = sg * 32;
            #pragma unroll 1
            for (int i = 0; i < 32; ++i) {
                const float val = s_sc[sq][kb0 + i];
                if (val > tv[TOPK - 1]) {
                    float cv = val;
                    int   ci = bat * SEQ + keybase + kb0 + i;  // global V row
                    #pragma unroll
                    for (int j = 0; j < TOPK; ++j) {
                        if (cv > tv[j]) {
                            float tf = tv[j]; tv[j] = cv; cv = tf;
                            int   tg = ti[j]; ti[j] = ci; ci = tg;
                        }
                    }
                }
            }
        }
        __syncthreads();
    }

    // ---- dump sorted candidate lists ----
    #pragma unroll
    for (int j = 0; j < TOPK; ++j) { s_mv[sq][sg][j] = tv[j]; s_mi[sq][sg][j] = ti[j]; }
    __syncthreads();

    // ---- 8-way merge of sorted lists + softmax (one thread per query) ----
    if (t < 16) {
        int heads[8] = {0, 0, 0, 0, 0, 0, 0, 0};
        float mx = 0.0f, sum = 0.0f;
        for (int n = 0; n < TOPK; ++n) {
            float best = -3.4e38f; int bs = 0;
            #pragma unroll
            for (int s2 = 0; s2 < 8; ++s2) {
                const int h = heads[s2];
                const float c = (h < TOPK) ? s_mv[t][s2][h] : -3.4e38f;
                if (c > best) { best = c; bs = s2; }
            }
            const int h = heads[bs];
            const int gi = s_mi[t][bs][h];
            heads[bs] = h + 1;
            if (n == 0) mx = best;                    // first pop is the max
            const float e = __expf(best - mx);
            sum += e;
            s_p [t][n] = e;
            s_pi[t][n] = gi;
        }
        const float inv = 1.0f / sum;
        for (int n = 0; n < TOPK; ++n) s_p[t][n] *= inv;
    }
    __syncthreads();

    // ---- phase 3: out[q, :] = sum_k p_k * V[idx_k, :]  (8 threads/query) ----
    {
        const int q2 = t >> 3;
        const int ds = (t & 7) * 32;
        float4 acc[8];
        #pragma unroll
        for (int j = 0; j < 8; ++j) acc[j] = make_float4(0.f, 0.f, 0.f, 0.f);
        #pragma unroll 1
        for (int kk = 0; kk < TOPK; ++kk) {
            const float p = s_p[q2][kk];
            const float4* vp = (const float4*)(vf + (size_t)s_pi[q2][kk] * DIM + ds);
            #pragma unroll
            for (int j = 0; j < 8; ++j) {
                const float4 vvv = vp[j];
                acc[j].x += p * vvv.x; acc[j].y += p * vvv.y;
                acc[j].z += p * vvv.z; acc[j].w += p * vvv.w;
            }
        }
        float4* op = (float4*)(out + (size_t)(q0 + q2) * DIM + ds);
        #pragma unroll
        for (int j = 0; j < 8; ++j) op[j] = acc[j];
    }
}

// ---------------------------------------------------------------------------
extern "C" void kernel_launch(void* const* d_in, const int* in_sizes, int n_in,
                              void* d_out, int out_size, void* d_ws, size_t ws_size,
                              hipStream_t stream) {
    const float* x  = (const float*)d_in[0];
    const float* Wq = (const float*)d_in[1];
    const float* bq = (const float*)d_in[2];
    const float* Wk = (const float*)d_in[3];
    const float* bk = (const float*)d_in[4];
    const float* Wv = (const float*)d_in[5];
    const float* bv = (const float*)d_in[6];
    float* out = (float*)d_out;

    const size_t rows = (size_t)BATCH * SEQ;             // 16384
    char* ws = (char*)d_ws;
    __bf16* qb = (__bf16*)ws;                            // 8 MiB
    __bf16* kb = (__bf16*)(ws + rows * DIM * 2);         // 8 MiB
    float*  vf = (float*) (ws + rows * DIM * 4);         // 16 MiB
    __bf16* Wqp = (__bf16*)(ws + rows * DIM * 8);                        // 128 KiB
    __bf16* Wkp = (__bf16*)(ws + rows * DIM * 8 + WPACK_FRAGS * 32);     // 128 KiB
    __bf16* Wvp = (__bf16*)(ws + rows * DIM * 8 + WPACK_FRAGS * 64);     // 128 KiB

    // Pack weights once into WMMA B-fragment layout (bf16).
    kmip_pack_w_kernel<<<WPACK_FRAGS / 256, 256, 0, stream>>>(Wq, Wqp);
    kmip_pack_w_kernel<<<WPACK_FRAGS / 256, 256, 0, stream>>>(Wk, Wkp);
    kmip_pack_w_kernel<<<WPACK_FRAGS / 256, 256, 0, stream>>>(Wv, Wvp);

    // 16384 output tiles / 4 waves per block
    kmip_qkv_kernel<<<4096, 128, 0, stream>>>(x, Wqp, bq, Wkp, bk, Wvp, bv, qb, kb, vf);
    // one block per (batch, 16-query tile)
    kmip_attn_kernel<<<BATCH * (SEQ / 16), 128, 0, stream>>>(qb, kb, vf, out);
}
